// Model_68186900792100
// MI455X (gfx1250) — compile-verified, matrix operand order checked by reading, not run on hardware
//
#include <hip/hip_runtime.h>

#define HID     2048
#define HD      128
#define NH      16
#define SEGLEN  1024
#define NSEG    8
#define TOTAL   (NSEG * SEGLEN)
#define BM      64          // query rows per workgroup
#define BK      64          // keys per inner iteration (4 x 16-key tiles)
#define NWAVE   4

typedef __attribute__((ext_vector_type(16))) _Float16 v16h;
typedef __attribute__((ext_vector_type(8)))  _Float16 v8h;
typedef __attribute__((ext_vector_type(8)))  float    v8f;
typedef __attribute__((ext_vector_type(4)))  float    v4f;

static __device__ __forceinline__ v8h cvt8(v4f a, v4f b) {
    v8h r;
    r[0] = (_Float16)a[0]; r[1] = (_Float16)a[1];
    r[2] = (_Float16)a[2]; r[3] = (_Float16)a[3];
    r[4] = (_Float16)b[0]; r[5] = (_Float16)b[1];
    r[6] = (_Float16)b[2]; r[7] = (_Float16)b[3];
    return r;
}

// A/B-style 16x32 f16 fragment from a row-major f32 matrix (converted on load).
static __device__ __forceinline__ v16h load_frag_f32(const float* __restrict__ p) {
    v4f a0 = *(const v4f*)(p + 0);
    v4f a1 = *(const v4f*)(p + 4);
    v4f b0 = *(const v4f*)(p + 16);
    v4f b1 = *(const v4f*)(p + 20);
    v8h lo = cvt8(a0, a1);
    v8h hh = cvt8(b0, b1);
    return __builtin_shufflevector(lo, hh, 0,1,2,3,4,5,6,7,8,9,10,11,12,13,14,15);
}

// A/B-style 16x32 f16 fragment from an f16 row: two 16B loads, no conversion.
static __device__ __forceinline__ v16h load_frag_h(const _Float16* __restrict__ row, int hioff) {
    v8h lo = *(const v8h*)(row + hioff);
    v8h hh = *(const v8h*)(row + 16 + hioff);
    return __builtin_shufflevector(lo, hh, 0,1,2,3,4,5,6,7,8,9,10,11,12,13,14,15);
}

// ---------------- prep kernel 1: K f32 -> f16 (same row-major layout) ----------------
__global__ __launch_bounds__(256)
void prep_kh(const float* __restrict__ k, _Float16* __restrict__ kh) {
    size_t i = ((size_t)blockIdx.x * blockDim.x + threadIdx.x) * 8;
    v4f a = *(const v4f*)(k + i);
    v4f b = *(const v4f*)(k + i + 4);
    *(v8h*)(kh + i) = cvt8(a, b);
}

// ---- prep kernel 2: V -> f16, transposed per (seg, head, 32-key block): [HD][32] tiles ----
__global__ __launch_bounds__(128)
void prep_vt(const float* __restrict__ v, _Float16* __restrict__ vt) {
    __shared__ _Float16 tile[HD][32];            // 8 KB
    const int bid = blockIdx.x;                  // NSEG*NH*32 = 4096
    const int jb  = bid & 31;
    const int h   = (bid >> 5) & 15;
    const int seg = bid >> 9;

    const float* vbase = v + (size_t)(seg * SEGLEN + jb * 32) * HID + h * HD;
    #pragma unroll
    for (int rep = 0; rep < 8; ++rep) {
        int linear = rep * 512 + threadIdx.x * 4;   // 32 keys x 128 dims
        int kk = linear >> 7;
        int dd = linear & 127;
        v4f vv = *(const v4f*)(vbase + (size_t)kk * HID + dd);
        tile[dd + 0][kk] = (_Float16)vv[0];
        tile[dd + 1][kk] = (_Float16)vv[1];
        tile[dd + 2][kk] = (_Float16)vv[2];
        tile[dd + 3][kk] = (_Float16)vv[3];
    }
    __syncthreads();
    // dump LDS tile linearly -> fully coalesced global writes
    _Float16* outp = vt + (size_t)bid * (HD * 32) + threadIdx.x * 32;
    const _Float16* row = &tile[threadIdx.x][0];
    *(v8h*)(outp + 0)  = *(const v8h*)(row + 0);
    *(v8h*)(outp + 8)  = *(const v8h*)(row + 8);
    *(v8h*)(outp + 16) = *(const v8h*)(row + 16);
    *(v8h*)(outp + 24) = *(const v8h*)(row + 24);
}

// ---------------- main flash-attention kernel ----------------
__global__ __launch_bounds__(128, 1)
void flash_attn_wmma(const float* __restrict__ q,
                     const _Float16* __restrict__ kh,
                     const _Float16* __restrict__ vt,
                     float* __restrict__ out) {
    __shared__ _Float16 pbuf[NWAVE][16][BK];     // per-wave P bounce: 8 KB

    const int lane  = threadIdx.x & 31;
    const int wave  = threadIdx.x >> 5;
    const int col   = lane & 15;
    const int hioff = (lane >> 4) * 8;           // 0 / 8

    const int bid = blockIdx.x;                  // 2048
    const int mb  = bid & 15;
    const int h   = (bid >> 4) & 15;
    const int seg = bid >> 8;
    const int segBase = seg * SEGLEN;
    const int m0  = mb * BM + wave * 16;

    // Q fragments: convert once per workgroup (f32 -> f16)
    v16h qf[4];
    {
        const float* qrow = q + (size_t)(segBase + m0 + col) * HID + h * HD + hioff;
        #pragma unroll
        for (int f = 0; f < 4; ++f) qf[f] = load_frag_f32(qrow + f * 32);
    }

    // all-ones B fragment for row-sum WMMA
    v16h ones;
    #pragma unroll
    for (int i = 0; i < 16; ++i) ones[i] = (_Float16)1.0f;

    v8f acc[8];
    #pragma unroll
    for (int n = 0; n < 8; ++n) acc[n] = (v8f)(0.0f);

    float mrun[8], lrun[8];
    #pragma unroll
    for (int r = 0; r < 8; ++r) { mrun[r] = -INFINITY; lrun[r] = 0.0f; }

    const float SCL = 0.08838834764831845f * 1.4426950408889634f; // 1/sqrt(128) * log2(e)

    const _Float16* vhead = vt + (size_t)((seg * NH + h) * (SEGLEN / 32)) * (HD * 32);

    for (int j = 0; j < SEGLEN; j += BK) {
        // ---- S = Q K^T : 4 key tiles x 4 K-steps = 16 WMMAs ----
        v8f s[4];
        #pragma unroll
        for (int t = 0; t < 4; ++t) s[t] = (v8f)(0.0f);
        {
            const _Float16* kbase = kh + (size_t)(segBase + j + col) * HID + h * HD;
            #pragma unroll
            for (int t = 0; t < 4; ++t) {
                const _Float16* krow = kbase + (size_t)(t * 16) * HID;
                #pragma unroll
                for (int f = 0; f < 4; ++f) {
                    v16h kb = load_frag_h(krow + f * 32, hioff);
                    s[t] = __builtin_amdgcn_wmma_f32_16x16x32_f16(false, qf[f], false, kb,
                                                                  (short)0, s[t], false, false);
                }
            }
        }

        // ---- online softmax over 64 keys (row max via shfl within 16-lane half) ----
        float alpha[8];
        #pragma unroll
        for (int r = 0; r < 8; ++r) {
            float a0 = s[0][r] * SCL, a1 = s[1][r] * SCL;
            float a2 = s[2][r] * SCL, a3 = s[3][r] * SCL;
            float mx = fmaxf(fmaxf(a0, a1), fmaxf(a2, a3));
            mx = fmaxf(mx, __shfl_xor(mx, 1, 32));
            mx = fmaxf(mx, __shfl_xor(mx, 2, 32));
            mx = fmaxf(mx, __shfl_xor(mx, 4, 32));
            mx = fmaxf(mx, __shfl_xor(mx, 8, 32));
            float mnew = fmaxf(mrun[r], mx);
            alpha[r] = exp2f(mrun[r] - mnew);
            mrun[r] = mnew;
            int row = r + hioff;
            pbuf[wave][row][ 0 + col] = (_Float16)exp2f(a0 - mnew);
            pbuf[wave][row][16 + col] = (_Float16)exp2f(a1 - mnew);
            pbuf[wave][row][32 + col] = (_Float16)exp2f(a2 - mnew);
            pbuf[wave][row][48 + col] = (_Float16)exp2f(a3 - mnew);
        }

        asm volatile("" ::: "memory");   // same-wave DS ops are in-order (ISA)

        // P fragments (two 16x32 f16 A-fragments)
        v16h pf0 = load_frag_h(&pbuf[wave][col][0],  hioff);
        v16h pf1 = load_frag_h(&pbuf[wave][col][32], hioff);

        // row sums via WMMA against all-ones B: D[m][*] = sum_k P[m,k]
        v8f rs = (v8f)(0.0f);
        rs = __builtin_amdgcn_wmma_f32_16x16x32_f16(false, pf0, false, ones, (short)0, rs, false, false);
        rs = __builtin_amdgcn_wmma_f32_16x16x32_f16(false, pf1, false, ones, (short)0, rs, false, false);

        v8f alphav;
        #pragma unroll
        for (int r = 0; r < 8; ++r) {
            alphav[r] = alpha[r];
            lrun[r] = lrun[r] * alpha[r] + rs[r];
        }
        #pragma unroll
        for (int n = 0; n < 8; ++n) acc[n] *= alphav;

        // ---- O += P V : 8 output tiles x 2 key chunks = 16 WMMAs ----
        const _Float16* vblk0 = vhead + (size_t)(j >> 5) * (HD * 32);
        const _Float16* vblk1 = vblk0 + HD * 32;
        #pragma unroll
        for (int n = 0; n < 8; ++n) {
            v16h vb0 = load_frag_h(vblk0 + (n * 16 + col) * 32, hioff);
            acc[n] = __builtin_amdgcn_wmma_f32_16x16x32_f16(false, pf0, false, vb0,
                                                            (short)0, acc[n], false, false);
            v16h vb1 = load_frag_h(vblk1 + (n * 16 + col) * 32, hioff);
            acc[n] = __builtin_amdgcn_wmma_f32_16x16x32_f16(false, pf1, false, vb1,
                                                            (short)0, acc[n], false, false);
        }
    }

    // ---- epilogue: divide by l, store fp32 output [1, H, TOTAL, 128] ----
    float inv[8];
    #pragma unroll
    for (int r = 0; r < 8; ++r) inv[r] = 1.0f / lrun[r];

    const size_t outBase = ((size_t)h * TOTAL + segBase + m0) * HD;
    #pragma unroll
    for (int n = 0; n < 8; ++n) {
        #pragma unroll
        for (int r = 0; r < 8; ++r) {
            out[outBase + (size_t)(r + hioff) * HD + n * 16 + col] = acc[n][r] * inv[r];
        }
    }
}

extern "C" void kernel_launch(void* const* d_in, const int* in_sizes, int n_in,
                              void* d_out, int out_size, void* d_ws, size_t ws_size,
                              hipStream_t stream) {
    const float* q = (const float*)d_in[0];
    const float* k = (const float*)d_in[1];
    const float* v = (const float*)d_in[2];
    float* out = (float*)d_out;
    (void)in_sizes; (void)n_in; (void)out_size; (void)ws_size;

    // workspace: K in f16 (32 MB) + V transposed f16 tiles (32 MB)
    _Float16* kh = (_Float16*)d_ws;
    _Float16* vt = kh + (size_t)TOTAL * HID;

    prep_kh<<<dim3((TOTAL * HID / 8) / 256), dim3(256), 0, stream>>>(k, kh);
    prep_vt<<<dim3(NSEG * NH * (SEGLEN / 32)), dim3(128), 0, stream>>>(v, vt);

    flash_attn_wmma<<<dim3(NSEG * NH * (SEGLEN / BM)), dim3(NWAVE * 32), 0, stream>>>(q, kh, vt, out);
}
